// stats_mtsnnMLP_63007170232445
// MI455X (gfx1250) — matrix operand order repaired and checked by reading, not compiled, not run on hardware
//
#include <hip/hip_runtime.h>

typedef __bf16 bf16_t;
typedef __attribute__((ext_vector_type(16))) __bf16 v16bf;
typedef __attribute__((ext_vector_type(8)))  __bf16 v8bf;
typedef __attribute__((ext_vector_type(4)))  __bf16 v4bf;
typedef __attribute__((ext_vector_type(8)))  float  v8f;
typedef __attribute__((ext_vector_type(4)))  float  v4f;

#define BATCH 4096
#define NIN   2048
#define NH    4096
#define NOUT  1024
#define LDSROW 40        // bf16 elems per LDS row: 32 data + 8 pad = 80B (16B aligned)
#define BUFELEMS (2 * 128 * LDSROW)   // one double-buffer stage: A tile + B tile

#define USE_ASYNC_LDS 1  // gfx1250 GLOBAL_LOAD_ASYNC_TO_LDS_B128 path (ASYNCcnt)

union FragBf { v16bf v; v8bf h[2]; };

// gfx1250 async copy: 16B per lane, global -> LDS, no VGPR staging (ASYNCcnt).
__device__ __forceinline__ void async_copy_b128(unsigned ldsOff, const void* g)
{
    unsigned long long ga = (unsigned long long)(size_t)g;
    asm volatile("global_load_async_to_lds_b128 %0, %1, off"
                 :: "v"(ldsOff), "v"(ga) : "memory");
}
__device__ __forceinline__ void wait_async4() {
    asm volatile("s_wait_asynccnt 0x4" ::: "memory");
}
__device__ __forceinline__ void wait_async0() {
    asm volatile("s_wait_asynccnt 0x0" ::: "memory");
}

// ---------------------------------------------------------------------------
// WMMA mainloop: C[128x128] tile of A(row-major MxK) x B(row-major NxK)^T.
// 256 threads = 8 waves arranged 4(M) x 2(N); each wave owns 32x64 = 2x4
// tiles of 16x16, accumulated over K in 32-wide slices. Double-buffered LDS
// fed by async global->LDS b128 copies (4 per thread per tile).
// ---------------------------------------------------------------------------
template<int K>
__device__ __forceinline__ void wmma_tile_k(const bf16_t* __restrict__ Ag,
                                            const bf16_t* __restrict__ Bg,
                                            int mBase, int nBase,
                                            bf16_t* __restrict__ lds,   // 2*BUFELEMS
                                            v8f acc[2][4])
{
    const int tid  = threadIdx.x;
    const int lane = tid & 31;
    const int wave = tid >> 5;
    const int l15  = lane & 15;
    const int hh   = lane >> 4;     // half-wave select
    const int wm   = wave & 3;      // wave M index (0..3)
    const int wn   = wave >> 2;     // wave N index (0..1)

    // Per-thread copy slots: 512 16B chunks per tile; this thread owns chunk
    // tid and tid+256 of both the A tile (128x32) and the B tile (128x32).
    const int c0 = tid, c1 = tid + 256;
    const int r0 = c0 >> 2, q0 = (c0 & 3) * 8;
    const int r1 = c1 >> 2, q1 = (c1 & 3) * 8;
    const bf16_t* gA0 = &Ag[(size_t)(mBase + r0) * K + q0];
    const bf16_t* gA1 = &Ag[(size_t)(mBase + r1) * K + q1];
    const bf16_t* gB0 = &Bg[(size_t)(nBase + r0) * K + q0];
    const bf16_t* gB1 = &Bg[(size_t)(nBase + r1) * K + q1];
    const unsigned dA0 = (unsigned)(size_t)&lds[r0 * LDSROW + q0];
    const unsigned dA1 = (unsigned)(size_t)&lds[r1 * LDSROW + q1];
    const unsigned dB0 = (unsigned)(size_t)&lds[128 * LDSROW + r0 * LDSROW + q0];
    const unsigned dB1 = (unsigned)(size_t)&lds[128 * LDSROW + r1 * LDSROW + q1];
    const unsigned bufB = (unsigned)(BUFELEMS * sizeof(bf16_t));

#if USE_ASYNC_LDS
    // Prologue: fill buffer 0 with the k=0 tiles.
    async_copy_b128(dA0, gA0);
    async_copy_b128(dA1, gA1);
    async_copy_b128(dB0, gB0);
    async_copy_b128(dB1, gB1);
#endif

    for (int k0 = 0; k0 < K; k0 += 32) {
        const int p = (k0 >> 5) & 1;
        const unsigned pOff  = p ? bufB : 0u;          // buffer being consumed
        const unsigned nOff  = p ? 0u   : bufB;        // buffer being filled
        const bool hasNext = (k0 + 32) < K;

#if USE_ASYNC_LDS
        if (hasNext) {
            const int kn = k0 + 32;
            async_copy_b128(dA0 + nOff, gA0 + kn);
            async_copy_b128(dA1 + nOff, gA1 + kn);
            async_copy_b128(dB0 + nOff, gB0 + kn);
            async_copy_b128(dB1 + nOff, gB1 + kn);
            wait_async4();   // this wave's current-buffer copies have landed
        } else {
            wait_async0();
        }
#else
        {
            bf16_t* ldsA = lds + (p ? BUFELEMS : 0);
            bf16_t* ldsB = ldsA + 128 * LDSROW;
            *(v8bf*)&ldsA[r0*LDSROW + q0] = *(const v8bf*)(gA0 + k0);
            *(v8bf*)&ldsA[r1*LDSROW + q1] = *(const v8bf*)(gA1 + k0);
            *(v8bf*)&ldsB[r0*LDSROW + q0] = *(const v8bf*)(gB0 + k0);
            *(v8bf*)&ldsB[r1*LDSROW + q1] = *(const v8bf*)(gB1 + k0);
        }
#endif
        __syncthreads();     // all waves' tile-p data visible in LDS

        const bf16_t* ldsA = lds + (p ? BUFELEMS : 0);
        const bf16_t* ldsB = ldsA + 128 * LDSROW;

        // A fragment (16x32 ISA layout): lane = row M=l15;
        //   e0..7 = K[8h..8h+7], e8..15 = K[16+8h..16+8h+7]
        FragBf a[2], b[4];
        #pragma unroll
        for (int i = 0; i < 2; ++i) {
            const int r = wm*32 + i*16 + l15;
            a[i].h[0] = *(const v8bf*)&ldsA[r*LDSROW + 8*hh];
            a[i].h[1] = *(const v8bf*)&ldsA[r*LDSROW + 16 + 8*hh];
        }
        // B fragment (32x16 ISA layout): lane = column N=l15;
        //   e0..7 = K[8h..8h+7], e8..15 = K[16+8h..16+8h+7]
        #pragma unroll
        for (int j = 0; j < 4; ++j) {
            const int r = wn*64 + j*16 + l15;
            b[j].h[0] = *(const v8bf*)&ldsB[r*LDSROW + 8*hh];
            b[j].h[1] = *(const v8bf*)&ldsB[r*LDSROW + 16 + 8*hh];
        }
        #pragma unroll
        for (int i = 0; i < 2; ++i)
            #pragma unroll
            for (int j = 0; j < 4; ++j)
                acc[i][j] = __builtin_amdgcn_wmma_f32_16x16x32_bf16(
                    false, a[i].v, false, b[j].v, (short)0, acc[i][j], false, false);

        __syncthreads();     // protect buffer p before it is refilled next iter
    }
}

// ---------------------------------------------------------------------------
// GEMM1: cur1 = x @ W1^T + b1    (done once; cur1 is reused every timestep)
// ---------------------------------------------------------------------------
__launch_bounds__(256)
__global__ void gemm1_kernel(const bf16_t* __restrict__ Xb,
                             const bf16_t* __restrict__ W1b,
                             const float*  __restrict__ b1,
                             float*        __restrict__ cur1)
{
    __shared__ __align__(16) bf16_t lds[2 * BUFELEMS];
    v8f acc[2][4] = {};
    const int mBase = blockIdx.y * 128;
    const int nBase = blockIdx.x * 128;
    wmma_tile_k<NIN>(Xb, W1b, mBase, nBase, lds, acc);

    const int tid = threadIdx.x, lane = tid & 31, wave = tid >> 5;
    const int l15 = lane & 15, hh = lane >> 4;
    const int wm = wave & 3, wn = wave >> 2;
    #pragma unroll
    for (int i = 0; i < 2; ++i) {
        const int rBase = mBase + wm*32 + i*16 + 8*hh;   // C layout: row = v + 8*(lane/16)
        #pragma unroll
        for (int j = 0; j < 4; ++j) {
            const int col  = nBase + wn*64 + j*16 + l15;
            const float bias = b1[col];
            #pragma unroll
            for (int v = 0; v < 8; ++v)
                cur1[(size_t)(rBase + v)*NH + col] = acc[i][j][v] + bias;
        }
    }
}

// ---------------------------------------------------------------------------
// LIF layer 1 (elementwise over [4096,4096]): mem1 update, spikes -> bf16,
// spike count -> wave32-reduced integer atomic.
// ---------------------------------------------------------------------------
__global__ void lif1_kernel(const float* __restrict__ cur1,
                            float*       __restrict__ mem1,
                            bf16_t*      __restrict__ spk1,
                            unsigned*    __restrict__ cnt1)
{
    const size_t i = (size_t)blockIdx.x * blockDim.x + threadIdx.x;  // 4 elems/thread
    v4f c = ((const v4f*)cur1)[i];
    v4f m = ((const v4f*)mem1)[i];
    v4bf s;
    unsigned local = 0;
    #pragma unroll
    for (int e = 0; e < 4; ++e) {
        float mm    = m[e];
        float reset = (mm > 1.0f) ? 1.0f : 0.0f;  // reset from incoming mem
        mm = 0.9f*mm + c[e] - reset;              // beta*mem + cur - reset*thresh
        m[e] = mm;
        const bool sp = mm > 1.0f;                // spike_fn(mem - 1)
        s[e] = sp ? (bf16_t)1.0f : (bf16_t)0.0f;
        local += sp ? 1u : 0u;
    }
    ((v4f*)mem1)[i] = m;
    ((v4bf*)spk1)[i] = s;
    const int lane = threadIdx.x & 31;
    #pragma unroll
    for (int off = 16; off > 0; off >>= 1) local += __shfl_xor(local, off, 32);
    if (lane == 0) atomicAdd(cnt1, local);
}

// ---------------------------------------------------------------------------
// GEMM2 fused with LIF layer 2: cur2 = spk1 @ W2^T + b2, then mem2 update,
// accumulate mem2 into d_out (for the mean over steps), count spikes.
// ---------------------------------------------------------------------------
__launch_bounds__(256)
__global__ void gemm2_lif2_kernel(const bf16_t* __restrict__ S1,
                                  const bf16_t* __restrict__ W2b,
                                  const float*  __restrict__ b2,
                                  float*        __restrict__ mem2,
                                  float*        __restrict__ out,
                                  unsigned*     __restrict__ cnt2)
{
    __shared__ __align__(16) bf16_t lds[2 * BUFELEMS];
    v8f acc[2][4] = {};
    const int mBase = blockIdx.y * 128;
    const int nBase = blockIdx.x * 128;
    wmma_tile_k<NH>(S1, W2b, mBase, nBase, lds, acc);

    const int tid = threadIdx.x, lane = tid & 31, wave = tid >> 5;
    const int l15 = lane & 15, hh = lane >> 4;
    const int wm = wave & 3, wn = wave >> 2;
    unsigned local = 0;
    #pragma unroll
    for (int i = 0; i < 2; ++i) {
        const int rBase = mBase + wm*32 + i*16 + 8*hh;
        #pragma unroll
        for (int j = 0; j < 4; ++j) {
            const int col  = nBase + wn*64 + j*16 + l15;
            const float bias = b2[col];
            #pragma unroll
            for (int v = 0; v < 8; ++v) {
                const size_t idx = (size_t)(rBase + v)*NOUT + col;
                const float c2 = acc[i][j][v] + bias;
                float m = mem2[idx];
                const float reset = (m > 1.0f) ? 1.0f : 0.0f;
                m = 0.9f*m + c2 - reset;
                mem2[idx] = m;
                out[idx] += m;                      // unique writer per element
                local += (m > 1.0f) ? 1u : 0u;
            }
        }
    }
    #pragma unroll
    for (int off = 16; off > 0; off >>= 1) local += __shfl_xor(local, off, 32);
    if (lane == 0) atomicAdd(cnt2, local);
}

// ---------------------------------------------------------------------------
// Utility kernels
// ---------------------------------------------------------------------------
__global__ void cvt_bf16_kernel(const float* __restrict__ in, bf16_t* __restrict__ out)
{
    const size_t i = (size_t)blockIdx.x * blockDim.x + threadIdx.x;
    v4f v = ((const v4f*)in)[i];
    v4bf o;
    #pragma unroll
    for (int e = 0; e < 4; ++e) o[e] = (bf16_t)v[e];
    ((v4bf*)out)[i] = o;
}

__global__ void zero4_kernel(float* __restrict__ p)
{
    const size_t i = (size_t)blockIdx.x * blockDim.x + threadIdx.x;
    v4f z = {0.f, 0.f, 0.f, 0.f};
    ((v4f*)p)[i] = z;
}

__global__ void zero_cnts_kernel(unsigned* __restrict__ c)
{
    if (threadIdx.x < 2) c[threadIdx.x] = 0u;
}

__global__ void finalize_kernel(float* __restrict__ out, const unsigned* __restrict__ cnts)
{
    const size_t i = (size_t)blockIdx.x * blockDim.x + threadIdx.x;
    v4f v = ((const v4f*)out)[i];
    #pragma unroll
    for (int e = 0; e < 4; ++e) v[e] *= 0.25f;    // mean over 4 timesteps
    ((v4f*)out)[i] = v;
    if (i == 0) {
        // overall_frequency = (cnt1 + cnt2) / (4 * B * (NH + NOUT))
        const double c = (double)cnts[0] + (double)cnts[1];
        out[(size_t)BATCH * NOUT] = (float)(c / (4.0 * BATCH * (NH + NOUT)));
    }
}

// ---------------------------------------------------------------------------
extern "C" void kernel_launch(void* const* d_in, const int* in_sizes, int n_in,
                              void* d_out, int out_size, void* d_ws, size_t ws_size,
                              hipStream_t stream)
{
    (void)in_sizes; (void)n_in; (void)out_size; (void)ws_size;
    const float* x  = (const float*)d_in[0];   // [4096,2048]
    const float* W1 = (const float*)d_in[1];   // [4096,2048]
    const float* b1 = (const float*)d_in[2];   // [4096]
    const float* W2 = (const float*)d_in[3];   // [1024,4096]
    const float* b2 = (const float*)d_in[4];   // [1024]
    float* out = (float*)d_out;                // [4096,1024] + 1 scalar

    char* w = (char*)d_ws;
    float*    cur1 = (float*)   (w + (size_t)0);          //  64 MB f32 [4096,4096]
    float*    mem1 = (float*)   (w + (size_t)67108864);   //  64 MB f32 [4096,4096]
    bf16_t*   spk1 = (bf16_t*)  (w + (size_t)134217728);  //  32 MB bf16 [4096,4096]
    float*    mem2 = (float*)   (w + (size_t)167772160);  //  16 MB f32 [4096,1024]
    bf16_t*   xb   = (bf16_t*)  (w + (size_t)184549376);  //  16 MB bf16 [4096,2048]
    bf16_t*   w1b  = (bf16_t*)  (w + (size_t)201326592);  //  16 MB bf16 [4096,2048]
    bf16_t*   w2b  = (bf16_t*)  (w + (size_t)218103808);  //   8 MB bf16 [1024,4096]
    unsigned* cnts = (unsigned*)(w + (size_t)226492416);  // 2 x u32 spike counters

    // State init (ws/d_out are poisoned; must be deterministic per call)
    zero4_kernel<<<16384, 256, 0, stream>>>(mem1);        // 16,777,216 f32
    zero4_kernel<<<4096,  256, 0, stream>>>(mem2);        //  4,194,304 f32
    zero4_kernel<<<4096,  256, 0, stream>>>(out);         //  4,194,304 f32 (mean acc)
    zero_cnts_kernel<<<1, 32, 0, stream>>>(cnts);

    // Precision convert for the WMMA path
    cvt_bf16_kernel<<<8192, 256, 0, stream>>>(x,  xb);
    cvt_bf16_kernel<<<8192, 256, 0, stream>>>(W1, w1b);
    cvt_bf16_kernel<<<4096, 256, 0, stream>>>(W2, w2b);

    // cur1 = x @ W1^T + b1  (once; M=4096, N=4096, K=2048)
    gemm1_kernel<<<dim3(32, 32), 256, 0, stream>>>(xb, w1b, b1, cur1);

    // 4 timesteps: LIF1 elementwise, then GEMM2 (M=4096,N=1024,K=4096) fused LIF2
    for (int s = 0; s < 4; ++s) {
        lif1_kernel<<<16384, 256, 0, stream>>>(cur1, mem1, spk1, cnts);
        gemm2_lif2_kernel<<<dim3(8, 32), 256, 0, stream>>>(spk1, w2b, b2, mem2, out, cnts + 1);
    }

    finalize_kernel<<<4096, 256, 0, stream>>>(out, cnts);
}
